// CoarseVAE_64269890617430
// MI455X (gfx1250) — matrix operand order repaired and checked by reading, not compiled, not run on hardware
//
#include <hip/hip_runtime.h>
#include <hip/hip_bf16.h>
#include <math.h>

// ---------------------------------------------------------------------------
// CoarseVAE pipeline for gfx1250 (MI455X).  GEMMs on v_wmma_f32_16x16x32_bf16
// (bf16 A/B, fp32 accumulate).  Edge-message MLPs computed per-node in LDS;
// neighbor-feature gathers use global_load_async_to_lds_b128 (ASYNCcnt path).
// ---------------------------------------------------------------------------

typedef __attribute__((ext_vector_type(16))) __bf16 v16bf;
typedef __attribute__((ext_vector_type(8)))  float  v8f;

#define NV     65536      // voxels
#define NB     32         // batches
#define MPTS   8192       // sampled points before extra
#define NPTS   10240      // batch_e size
#define KNN    32         // neighbors
#define FEAT   256
#define LAT    256

// ---------------- helpers ---------------------------------------------------

__device__ __forceinline__ unsigned short f2bf(float x) {
  union { float f; unsigned u; } c; c.f = x;
  unsigned r = (c.u + 0x7FFFu + ((c.u >> 16) & 1u)) >> 16;   // RNE f32->bf16
  return (unsigned short)r;
}

// A-matrix (16x32 bf16) lane/element -> K index   (ISA 7.12.2, wave32)
__device__ __forceinline__ int a_k(int lane, int e) {
  return ((e < 8) ? e : e + 8) + ((lane >= 16) ? 8 : 0);
}
// B-matrix (32x16 bf16) lane/element -> K index
__device__ __forceinline__ int b_k(int lane, int e) {
  return ((e < 8) ? e : e + 8) + ((lane >= 16) ? 8 : 0);
}

template <typename FA, typename FB>
__device__ __forceinline__ void wmma_bf16(v8f& acc, FA fa, FB fb) {
  union { v16bf v; unsigned short s[16]; } A, B;
#pragma unroll
  for (int e = 0; e < 16; ++e) { A.s[e] = fa(e); B.s[e] = fb(e); }
  acc = __builtin_amdgcn_wmma_f32_16x16x32_bf16(
      /*neg_a=*/false, A.v, /*neg_b=*/false, B.v,
      /*c_mod=*/(short)0, acc, /*reuse_a=*/false, /*reuse_b=*/false);
}

// per-lane async gather: 16B  global -> LDS   (tracked by ASYNCcnt)
__device__ __forceinline__ void async_g2l_b128(unsigned lds_off,
                                               unsigned long long gaddr) {
  asm volatile("global_load_async_to_lds_b128 %0, %1, off"
               :: "v"(lds_off), "v"(gaddr) : "memory");
}
__device__ __forceinline__ void wait_async0() {
  asm volatile("s_wait_asynccnt 0x0" ::: "memory");
}

__device__ __forceinline__ unsigned pcg(unsigned x) {
  x = x * 747796405u + 2891336453u;
  unsigned w = ((x >> ((x >> 28) + 4u)) ^ x) * 277803737u;
  return (w >> 22) ^ w;
}
__device__ __forceinline__ float gauss(unsigned seed, unsigned i) {
  unsigned h1 = pcg(seed ^ pcg(i + 0x9E3779B9u));
  unsigned h2 = pcg(h1 + 0x85EBCA6Bu);
  float u1 = ((h1 >> 8) + 0.5f) * (1.0f / 16777216.0f);
  float u2 = ((h2 >> 8) + 0.5f) * (1.0f / 16777216.0f);
  return sqrtf(-2.0f * logf(u1)) * cosf(6.28318530718f * u2);
}

// ---------------- small utility kernels -------------------------------------

__global__ void k_zero(float* p, int n) {
  int i = blockIdx.x * blockDim.x + threadIdx.x;
  if (i < n) p[i] = 0.f;
}

__global__ void k_seg_pos(const float* __restrict__ pos, const int* __restrict__ vb,
                          float* possum, float* cnt) {
  int i = blockIdx.x * blockDim.x + threadIdx.x;
  if (i >= NV) return;
  int b = vb[i];
  atomicAdd(&possum[b * 3 + 0], pos[i * 3 + 0]);
  atomicAdd(&possum[b * 3 + 1], pos[i * 3 + 1]);
  atomicAdd(&possum[b * 3 + 2], pos[i * 3 + 2]);
  atomicAdd(&cnt[b], 1.0f);
}

__global__ void k_center(const float* possum, const float* cnt, float* center) {
  int b = threadIdx.x;
  if (b >= NB) return;
  float c = cnt[b]; c = c < 1.f ? 1.f : c;
  for (int d = 0; d < 3; ++d) center[b * 3 + d] = possum[b * 3 + d] / c;
}

// ---------------- encoder: h = relu([feat|rel] @ Wl + bl), fused segment-sum -

__global__ __launch_bounds__(128)
void k_encoder(const float* __restrict__ feat, const float* __restrict__ pos,
               const int* __restrict__ vb, const float* __restrict__ center,
               const float* __restrict__ Wl, const float* __restrict__ bl,
               float* hsum) {
  const int lane = threadIdx.x & 31, wave = threadIdx.x >> 5;
  const int row0 = blockIdx.x * 16;
  const int m    = row0 + (lane & 15);     // A-fetch row for this lane
  const int bidx = vb[m];

  for (int cp = 0; cp < 4; ++cp) {         // 4 passes of 64 cols (4 waves x 16)
    const int col0 = (cp * 4 + wave) * 16;
    const int col  = col0 + (lane & 15);
    v8f acc = {};
    for (int kt = 0; kt < 9; ++kt) {       // K = 259 padded to 288
      const int kk = kt * 32;
      wmma_bf16(acc,
        [&](int e) -> unsigned short {
          int k = kk + a_k(lane, e);
          float v = 0.f;
          if (k < FEAT)            v = feat[(size_t)m * FEAT + k];
          else if (k < FEAT + 3)   v = pos[m * 3 + (k - FEAT)] - center[bidx * 3 + (k - FEAT)];
          return f2bf(v);
        },
        [&](int e) -> unsigned short {
          int k = kk + b_k(lane, e);
          return (k < FEAT + 3) ? f2bf(Wl[(size_t)k * 256 + col]) : (unsigned short)0;
        });
    }
    union { v8f v; float f[8]; } c; c.v = acc;
    const float bias = bl[col];
    const int rbase = row0 + ((lane >= 16) ? 8 : 0);
#pragma unroll
    for (int r = 0; r < 8; ++r) {
      float v = c.f[r] + bias;
      v = v > 0.f ? v : 0.f;                       // ReLU
      atomicAdd(&hsum[vb[rbase + r] * 256 + col], v);  // fused segment-sum
    }
  }
}

__global__ void k_agg(const float* hsum, const float* cnt, float* agg) {
  int i = blockIdx.x * blockDim.x + threadIdx.x;
  if (i >= NB * 256) return;
  float c = cnt[i >> 8]; c = c < 1.f ? 1.f : c;
  agg[i] = hsum[i] / c;
}

// ---------------- latent GEMM: out[32x512] = agg @ Wg + bg ------------------
// agg (32 KB) is staged into LDS with async b128 loads and reused by all waves.

__global__ __launch_bounds__(256)
void k_latent(const float* __restrict__ agg, const float* __restrict__ Wg,
              const float* __restrict__ bg, float* lat) {
  __shared__ __align__(16) float sA[32 * 256];
  for (int t = threadIdx.x; t < 512; t += 256) {   // 512 x 16B chunks
    unsigned lds_off = (unsigned)(unsigned long long)&sA[t * 4];
    async_g2l_b128(lds_off, (unsigned long long)(agg + t * 4));
  }
  wait_async0();
  __syncthreads();

  const int lane = threadIdx.x & 31, wave = threadIdx.x >> 5;
  for (int t = wave; t < 64; t += 8) {     // 2 row-tiles x 32 col-tiles
    const int mt = t >> 5, nt = t & 31;
    const int col = nt * 16 + (lane & 15);
    const int arow = mt * 16 + (lane & 15);
    v8f acc = {};
    for (int kk = 0; kk < 256; kk += 32) {
      wmma_bf16(acc,
        [&](int e) -> unsigned short {
          int k = kk + a_k(lane, e);
          return f2bf(sA[arow * 256 + k]);
        },
        [&](int e) -> unsigned short {
          int k = kk + b_k(lane, e);
          return f2bf(Wg[(size_t)k * 512 + col]);
        });
    }
    union { v8f v; float f[8]; } c; c.v = acc;
    const float bias = bg[col];
#pragma unroll
    for (int r = 0; r < 8; ++r) {
      int row = mt * 16 + r + ((lane >= 16) ? 8 : 0);
      lat[row * 512 + col] = c.f[r] + bias;
    }
  }
}

__global__ void k_latent_post(const float* lat, float* out_mu, float* out_sig, float* z) {
  int i = blockIdx.x * blockDim.x + threadIdx.x;
  if (i >= NB * 256) return;
  int row = i >> 8, c = i & 255;
  float mu = lat[row * 512 + c];
  float sr = lat[row * 512 + 256 + c];
  float sig = (sr > 20.f ? sr : log1pf(expf(sr))) + 1e-4f;  // softplus + 1e-4
  out_mu[i] = mu;
  out_sig[i] = sig;
  z[i] = mu + sig * gauss(1u, (unsigned)i);                 // reparameterize
}

// ---------------- point population + batch_e --------------------------------

__global__ void k_make_pts(const int* __restrict__ batch, int* be, float* be_out,
                           float* pop) {
  int i = blockIdx.x * blockDim.x + threadIdx.x;
  if (i >= NPTS) return;
  int b;
  if (i < MPTS) b = batch[i];
  else {  // deterministic no-replacement subsample (odd stride perm of [0,8192))
    unsigned j = (4099u * (unsigned)(i - MPTS) + 17u) & (MPTS - 1);
    b = batch[j];
  }
  be[i] = b;
  be_out[i] = (float)b;
  for (int d = 0; d < 3; ++d)
    pop[i * 3 + d] = 0.3f * gauss(3u, (unsigned)(i * 3 + d));
}

// ---------------- radius-KNN (all pairs, same batch, top-K smallest d2) -----

__global__ __launch_bounds__(128)
void k_knn(const float* __restrict__ pop, const int* __restrict__ be, float r2,
           int* __restrict__ idxo, int* __restrict__ cnto) {
  int i = blockIdx.x * blockDim.x + threadIdx.x;
  if (i >= NPTS) return;
  const float px = pop[i * 3], py = pop[i * 3 + 1], pz = pop[i * 3 + 2];
  const int b = be[i];
  float bd[KNN]; int bi[KNN];
#pragma unroll
  for (int t = 0; t < KNN; ++t) { bd[t] = 3.4e38f; bi[t] = -1; }
  int filled = 0;
  for (int j = 0; j < NPTS; ++j) {
    if ((j & 63) == 0)  // streaming hint: pull the next tile toward the WGP
      __builtin_prefetch(&pop[(j + 64 < NPTS ? j + 64 : j) * 3], 0, 3);
    if (j == i || be[j] != b) continue;
    float dx = pop[j * 3] - px, dy = pop[j * 3 + 1] - py, dz = pop[j * 3 + 2] - pz;
    float d2 = dx * dx + dy * dy + dz * dz;
    if (d2 > r2 || d2 >= bd[KNN - 1]) continue;
    int t = KNN - 1;
    while (t > 0 && bd[t - 1] > d2) { bd[t] = bd[t - 1]; bi[t] = bi[t - 1]; --t; }
    bd[t] = d2; bi[t] = j;
    if (filled < KNN) ++filled;
  }
  for (int t = 0; t < KNN; ++t) idxo[i * KNN + t] = bi[t];
  cnto[i] = filled;
}

// ---------------- ar_layer: per-node edge-MLP + mean + g-MLP ----------------
// One workgroup (4 waves) per node.  Neighbor rows gathered into LDS with
// async b128 loads; f-MLP via bf16 WMMA; masked mean fused into the GEMM2
// epilogue via LDS float atomics; then per-node g-MLP.

template <int XD, int FH, int FO, int GH, int GO, int LAYER>
__global__ __launch_bounds__(128)
void k_ar(const float* __restrict__ xin, const float* __restrict__ zbuf,
          const int* __restrict__ be, const float* __restrict__ popin,
          const int* __restrict__ idx, const int* __restrict__ cnt,
          const float* __restrict__ Wf1, const float* __restrict__ bf1,
          const float* __restrict__ Wf2, const float* __restrict__ bf2,
          const float* __restrict__ Wg1, const float* __restrict__ bg1,
          const float* __restrict__ Wg2, const float* __restrict__ bg2,
          float* __restrict__ fout, float* __restrict__ popout) {
  constexpr int MSGD = XD + 3;
  constexpr int KP1  = ((MSGD + 31) / 32) * 32;   // padded K for GEMM1
  constexpr int KP2  = ((FH + 31) / 32) * 32;     // padded K for GEMM2
  constexpr int NT1  = FH / 16;
  constexpr int NT2  = FO / 16;
  constexpr int XDA  = (LAYER == 1) ? LAT : XD;   // async-copied row prefix
  constexpr int CHK  = XDA / 4;                   // b128 chunks per row

  __shared__ __align__(16) float sMsg[32 * KP1];  // fp32 message tile
  __shared__ float sH1[32 * FH];
  __shared__ float sAgg[FO];
  __shared__ float sG[GH];

  const int i = blockIdx.x;
  const int lane = threadIdx.x & 31, wave = threadIdx.x >> 5;
  const int ci = cnt[i];

  // ---- stage 1a: async gather of neighbor feature rows into LDS ----
  for (int c = threadIdx.x; c < FO; c += 128) sAgg[c] = 0.f;
  for (int t = threadIdx.x; t < 32 * CHK; t += 128) {
    int e = t / CHK, c = t - e * CHK;
    int j = idx[i * KNN + e];
    if (j < 0) j = i;                       // invalid edge: safe gather, masked later
    const float* src = (LAYER == 1) ? (zbuf + (size_t)be[j] * LAT + c * 4)
                                    : (xin + (size_t)j * XD + c * 4);
    unsigned lds_off = (unsigned)(unsigned long long)&sMsg[e * KP1 + c * 4];
    async_g2l_b128(lds_off, (unsigned long long)src);
  }
  // ---- stage 1b: scalar tail (pop part / rel / zero pad), overlapped ----
  for (int t = threadIdx.x; t < 32 * (KP1 - XDA); t += 128) {
    int e = t / (KP1 - XDA), m = t - e * (KP1 - XDA);
    int k = XDA + m;
    int j = idx[i * KNN + e];
    if (j < 0) j = i;
    float v = 0.f;
    if (k < XD)           v = popin[j * 3 + (k - LAT)];  // LAYER1: pop part of x
    else if (k < MSGD)    v = popin[j * 3 + (k - XD)] - popin[i * 3 + (k - XD)];
    sMsg[e * KP1 + k] = v;
  }
  wait_async0();
  __syncthreads();

  // ---- stage 2: H1 = relu(Msg @ Wf1 + bf1) ----
  for (int t = wave; t < 2 * NT1; t += 4) {
    const int mt = t / NT1, nt = t % NT1;
    const int col = nt * 16 + (lane & 15);
    const int arow = mt * 16 + (lane & 15);
    v8f acc = {};
    for (int kk = 0; kk < KP1; kk += 32) {
      wmma_bf16(acc,
        [&](int e) -> unsigned short {
          return f2bf(sMsg[arow * KP1 + kk + a_k(lane, e)]);
        },
        [&](int e) -> unsigned short {
          int k = kk + b_k(lane, e);
          return (k < MSGD) ? f2bf(Wf1[(size_t)k * FH + col]) : (unsigned short)0;
        });
    }
    union { v8f v; float f[8]; } c; c.v = acc;
    const float bias = bf1[col];
#pragma unroll
    for (int r = 0; r < 8; ++r) {
      int row = mt * 16 + r + ((lane >= 16) ? 8 : 0);
      float v = c.f[r] + bias;
      sH1[row * FH + col] = v > 0.f ? v : 0.f;
    }
  }
  __syncthreads();

  // ---- stage 3: H2 = H1 @ Wf2 + bf2, masked mean fused into epilogue ----
  for (int t = wave; t < 2 * NT2; t += 4) {
    const int mt = t / NT2, nt = t % NT2;
    const int col = nt * 16 + (lane & 15);
    const int arow = mt * 16 + (lane & 15);
    v8f acc = {};
    for (int kk = 0; kk < KP2; kk += 32) {
      wmma_bf16(acc,
        [&](int e) -> unsigned short {
          int k = kk + a_k(lane, e);
          return (k < FH) ? f2bf(sH1[arow * FH + k]) : (unsigned short)0;
        },
        [&](int e) -> unsigned short {
          int k = kk + b_k(lane, e);
          return (k < FH) ? f2bf(Wf2[(size_t)k * FO + col]) : (unsigned short)0;
        });
    }
    union { v8f v; float f[8]; } c; c.v = acc;
    const float bias = bf2[col];
    float partial = 0.f;
#pragma unroll
    for (int r = 0; r < 8; ++r) {
      int row = mt * 16 + r + ((lane >= 16) ? 8 : 0);
      if (row < ci) partial += c.f[r] + bias;   // mask invalid edges
    }
    atomicAdd(&sAgg[col], partial);             // ds_add_f32 (LDS atomic)
  }
  __syncthreads();

  // ---- stage 4: g-MLP (per-node vectors; VALU dot products) ----
  const float inv = 1.0f / (float)(ci < 1 ? 1 : ci);
  for (int g = threadIdx.x; g < GH; g += 128) {
    float s = bg1[g];
    for (int c = 0; c < FO; ++c) s += (sAgg[c] * inv) * Wg1[(size_t)c * GH + g];
    sG[g] = s > 0.f ? s : 0.f;
  }
  __syncthreads();
  for (int o = threadIdx.x; o < GO; o += 128) {
    float s = bg2[o];
    for (int g = 0; g < GH; ++g) s += sG[g] * Wg2[(size_t)g * GO + o];
    constexpr int FDIM = GO - 3;
    if (o < FDIM) {
      fout[(size_t)i * FDIM + o] = s > 0.f ? s : (expf(s) - 1.0f);  // celu
    } else {
      popout[i * 3 + (o - FDIM)] = s;
    }
  }
}

// ---------------- host-side orchestration -----------------------------------

extern "C" void kernel_launch(void* const* d_in, const int* in_sizes, int n_in,
                              void* d_out, int out_size, void* d_ws, size_t ws_size,
                              hipStream_t stream) {
  // Input layout: [voxel__pos, voxel__feature, <params pytree-flattened,
  // dict keys alphabetical: ar1{Wf1,Wf2,Wg1,Wg2,bf1,bf2,bg1,bg2}, ar2{..},
  // ar3{..}, pc_Wg, pc_bg, pc_Wl, pc_bl>, voxel__batch, batch]
  const float* vpos  = (const float*)d_in[0];
  const float* vfeat = (const float*)d_in[1];
  const float* P[28];
  for (int t = 0; t < 28; ++t) P[t] = (const float*)d_in[2 + t];
  const float *a1Wf1=P[0],*a1Wf2=P[1],*a1Wg1=P[2],*a1Wg2=P[3],*a1bf1=P[4],*a1bf2=P[5],*a1bg1=P[6],*a1bg2=P[7];
  const float *a2Wf1=P[8],*a2Wf2=P[9],*a2Wg1=P[10],*a2Wg2=P[11],*a2bf1=P[12],*a2bf2=P[13],*a2bg1=P[14],*a2bg2=P[15];
  const float *a3Wf1=P[16],*a3Wf2=P[17],*a3Wg1=P[18],*a3Wg2=P[19],*a3bf1=P[20],*a3bf2=P[21],*a3bg1=P[22],*a3bg2=P[23];
  const float *pcWg=P[24],*pcbg=P[25],*pcWl=P[26],*pcbl=P[27];
  const int* vbatch = (const int*)d_in[30];
  const int* batch  = (const int*)d_in[31];

  // workspace layout (float elements)
  float* ws = (float*)d_ws;
  float* possum = ws + 0;            // 96
  float* cntb   = ws + 96;           // 32
  float* hsum   = ws + 128;          // 8192
  float* center = ws + 8320;         // 96
  float* agg    = ws + 8416;         // 8192
  float* lat    = ws + 16608;        // 16384
  float* zbuf   = ws + 32992;        // 8192
  int*   be     = (int*)(ws + 41184);// 10240
  float* popa   = ws + 51424;        // 30720
  float* popb   = ws + 82144;        // 30720
  float* f64b   = ws + 112864;       // 655360
  float* f32b   = ws + 768224;       // 327680
  int*   kidx   = (int*)(ws + 1095904); // 327680
  int*   kcnt   = (int*)(ws + 1423584); // 10240

  float* out_pop = (float*)d_out;             // NPTS*3
  float* out_be  = out_pop + NPTS * 3;        // NPTS
  float* out_mu  = out_be + NPTS;             // 8192
  float* out_sig = out_mu + NB * 256;         // 8192

  // 0) zero atomic accumulators
  k_zero<<<(8320 + 255) / 256, 256, 0, stream>>>(ws, 8320);
  // 1) per-batch position mean
  k_seg_pos<<<NV / 256, 256, 0, stream>>>(vpos, vbatch, possum, cntb);
  k_center<<<1, 32, 0, stream>>>(possum, cntb, center);
  // 2) encoder GEMM + fused segment-sum (WMMA)
  k_encoder<<<NV / 16, 128, 0, stream>>>(vfeat, vpos, vbatch, center, pcWl, pcbl, hsum);
  k_agg<<<(NB * 256 + 255) / 256, 256, 0, stream>>>(hsum, cntb, agg);
  // 3) latent GEMM (WMMA, async-staged A) + softplus/reparam
  k_latent<<<1, 256, 0, stream>>>(agg, pcWg, pcbg, lat);
  k_latent_post<<<(NB * 256 + 255) / 256, 256, 0, stream>>>(lat, out_mu, out_sig, zbuf);
  // 4) batch_e + point population
  k_make_pts<<<NPTS / 128, 128, 0, stream>>>(batch, be, out_be, popa);
  // 5) ar1
  k_knn<<<NPTS / 128, 128, 0, stream>>>(popa, be, 0.2f * 0.2f, kidx, kcnt);
  k_ar<259, 128, 128, 128, 67, 1><<<NPTS, 128, 0, stream>>>(
      zbuf, zbuf, be, popa, kidx, kcnt,
      a1Wf1, a1bf1, a1Wf2, a1bf2, a1Wg1, a1bg1, a1Wg2, a1bg2, f64b, popb);
  // 6) ar2
  k_knn<<<NPTS / 128, 128, 0, stream>>>(popb, be, 0.1f * 0.1f, kidx, kcnt);
  k_ar<64, 64, 64, 64, 35, 2><<<NPTS, 128, 0, stream>>>(
      f64b, zbuf, be, popb, kidx, kcnt,
      a2Wf1, a2bf1, a2Wf2, a2bf2, a2Wg1, a2bg1, a2Wg2, a2bg2, f32b, popa);
  // 7) ar3 -> final pop straight into d_out
  k_knn<<<NPTS / 128, 128, 0, stream>>>(popa, be, 0.05f * 0.05f, kidx, kcnt);
  k_ar<32, 16, 16, 16, 3, 3><<<NPTS, 128, 0, stream>>>(
      f32b, zbuf, be, popa, kidx, kcnt,
      a3Wf1, a3bf1, a3Wf2, a3bf2, a3Wg1, a3bg1, a3Wg2, a3bg2, f32b, out_pop);

  (void)in_sizes; (void)n_in; (void)out_size; (void)ws_size;
}
